// CasualSelfAttention_2001454759997
// MI455X (gfx1250) — compile-verified, matrix operand order checked by reading, not run on hardware
//
#include <hip/hip_runtime.h>
#include <hip/hip_bf16.h>
#include <stdint.h>

// ---------------------------------------------------------------------------
// Causal self-attention block for MI455X (gfx1250, wave32, WMMA + TDM).
//   B=4, T=2048, C=768, H=12, hd=64.  fp32 in/out; bf16 WMMA compute with
//   fp32 accumulation; bf16 intermediates in d_ws (qkv 37.7 MB, yatt 12.6 MB).
// Pipeline: [QKV GEMM fp32->bf16] -> [flash attention, TDM-staged K/Q tiles,
//           ones-WMMA row sums] -> [output GEMM bf16->fp32].
// ---------------------------------------------------------------------------

typedef __attribute__((ext_vector_type(16))) __bf16   v16bf;
typedef __attribute__((ext_vector_type(8)))  __bf16   v8bf;
typedef __attribute__((ext_vector_type(4)))  __bf16   v4bf;
typedef __attribute__((ext_vector_type(8)))  float    v8f;
typedef __attribute__((ext_vector_type(4)))  float    v4f;
typedef __attribute__((ext_vector_type(4)))  unsigned int v4u;
typedef __attribute__((ext_vector_type(8)))  int      v8i;
typedef __attribute__((ext_vector_type(4)))  int      v4i;

#define WMMA_BF16(a, b, c) \
  __builtin_amdgcn_wmma_f32_16x16x32_bf16(false, (a), false, (b), (short)0, (c), false, false)

static __device__ __forceinline__ v16bf cat16(v8bf lo, v8bf hi) {
  v16bf r;
#pragma unroll
  for (int i = 0; i < 8; ++i) { r[i] = lo[i]; r[i + 8] = hi[i]; }
  return r;
}

// ---------------------------------------------------------------------------
// TDM: async 2D bf16 tile copy global->LDS (D# per CDNA5 ISA).
// group0: count=1 | lds_addr | global_addr(57b) | type=2 (image)
// group1: data_size=1(2B), tensor_dim0/1, tile_dim0/1, tensor_dim0_stride
// 6-arg builtin (clang-23 / therock-10.0):
//   (uint32x4 g0, int32x8 g1, int32x4, int32x4, int32x8, i32 cpol)
// Groups 2/3 disabled (2D tensor) -> zero-filled trailing operands.
// Tracked by TENSORcnt; issuer must s_wait_tensorcnt before publishing.
// ---------------------------------------------------------------------------
static __device__ __forceinline__ void tdm_load_2d_bf16(
    unsigned int lds_addr, const __bf16* gptr,
    unsigned int tensor_w, unsigned int tensor_h,
    unsigned int stride_elems, unsigned int tile_w, unsigned int tile_h) {
  unsigned long long ga = (unsigned long long)(uintptr_t)gptr;
  v4u g0;
  g0[0] = 1u;                                                  // count=1
  g0[1] = lds_addr;                                            // lds_addr
  g0[2] = (unsigned int)ga;                                    // global_addr lo
  g0[3] = (unsigned int)((ga >> 32) & 0x1FFFFFFu) | (2u << 30);// addr hi | type=2
  v8i g1;
  g1[0] = (int)(1u << 16);                       // wg_mask=0, data_size=1 (2B)
  g1[1] = (int)(tensor_w << 16);                 // tensor_dim0[15:0]
  g1[2] = (int)((tensor_w >> 16) | (tensor_h << 16));
  g1[3] = (int)((tensor_h >> 16) | (tile_w << 16));
  g1[4] = (int)(tile_h & 0xFFFFu);               // tile_dim1 (tile_dim2=0)
  g1[5] = (int)stride_elems;                     // tensor_dim0_stride lo
  g1[6] = 0;                                     // stride hi / dim1_stride
  g1[7] = 0;
  v4i z4 = {0, 0, 0, 0};
  v8i z8 = {0, 0, 0, 0, 0, 0, 0, 0};
  __builtin_amdgcn_tensor_load_to_lds(g0, g1, z4, z4, z8, 0);
}

template <typename TC> static __device__ __forceinline__ TC cvt_out(float v);
template <> __device__ __forceinline__ float  cvt_out<float >(float v) { return v; }
template <> __device__ __forceinline__ __bf16 cvt_out<__bf16>(float v) { return (__bf16)v; }

// A-tile register staging (128x32 elements, 256 threads), fp32 or bf16 source.
template <typename TA> struct AStage;
template <> struct AStage<float> {
  v4f r[4];
  __device__ __forceinline__ void fetch(const float* A, int m0, int K, int k0, int tid) {
#pragma unroll
    for (int i = 0; i < 4; ++i) {
      int flat = i * 1024 + tid * 4;
      r[i] = *(const v4f*)(A + (size_t)(m0 + (flat >> 5)) * K + k0 + (flat & 31));
    }
  }
  __device__ __forceinline__ void store(__bf16* sA, int tid) {
#pragma unroll
    for (int i = 0; i < 4; ++i) {
      int flat = i * 1024 + tid * 4;
      v4bf ab;
      ab[0] = (__bf16)r[i][0]; ab[1] = (__bf16)r[i][1];
      ab[2] = (__bf16)r[i][2]; ab[3] = (__bf16)r[i][3];
      *(v4bf*)&sA[(flat >> 5) * 32 + (flat & 31)] = ab;
    }
  }
};
template <> struct AStage<__bf16> {
  v8bf r[2];
  __device__ __forceinline__ void fetch(const __bf16* A, int m0, int K, int k0, int tid) {
#pragma unroll
    for (int i = 0; i < 2; ++i) {
      int flat = i * 2048 + tid * 8;
      r[i] = *(const v8bf*)(A + (size_t)(m0 + (flat >> 5)) * K + k0 + (flat & 31));
    }
  }
  __device__ __forceinline__ void store(__bf16* sA, int tid) {
#pragma unroll
    for (int i = 0; i < 2; ++i) {
      int flat = i * 2048 + tid * 8;
      *(v8bf*)&sA[(flat >> 5) * 32 + (flat & 31)] = r[i];
    }
  }
};

// ---------------------------------------------------------------------------
// GEMM: C[M,N] = A[M,K] * W[K,N](fp32) + bias[N].  Block tile 128x128, BK=32,
// 256 threads = 8 waves (4x2), wave tile 32x64 = 2x4 WMMA.  Double-buffered
// LDS: fetch tile k+1 to regs -> WMMA on tile k -> convert/store tile k+1.
// ---------------------------------------------------------------------------
template <typename TA, typename TC>
__global__ __launch_bounds__(256) void gemm_bias_kernel(
    const TA* __restrict__ A, const float* __restrict__ W,
    const float* __restrict__ bias, TC* __restrict__ C,
    int M, int N, int K) {
  __shared__ __align__(128) __bf16 sA [2][128 * 32];  // [buf][row][k] 16 KB
  __shared__ __align__(128) __bf16 sBT[2][128 * 32];  // [buf][n][k]   16 KB

  const int tid  = threadIdx.x;
  const int lane = tid & 31, wid = tid >> 5;
  const int wm = wid & 3, wn = wid >> 2;
  const int m0 = blockIdx.y * 128, n0 = blockIdx.x * 128;
  const int kb = (lane >> 4) * 8, nlo = lane & 15;

  v8f acc[2][4] = {};
  AStage<TA> ast;
  v4f wr[4];

  // prologue: tile 0 -> LDS buffer 0
  ast.fetch(A, m0, K, 0, tid);
#pragma unroll
  for (int i = 0; i < 4; ++i) {
    int flat = i * 1024 + tid * 4;
    wr[i] = *(const v4f*)(W + (size_t)(flat >> 7) * N + n0 + (flat & 127));
  }
  ast.store(sA[0], tid);
#pragma unroll
  for (int i = 0; i < 4; ++i) {
    int flat = i * 1024 + tid * 4;
    int nn = flat & 127, kk = flat >> 7;
    sBT[0][(nn + 0) * 32 + kk] = (__bf16)wr[i][0];
    sBT[0][(nn + 1) * 32 + kk] = (__bf16)wr[i][1];
    sBT[0][(nn + 2) * 32 + kk] = (__bf16)wr[i][2];
    sBT[0][(nn + 3) * 32 + kk] = (__bf16)wr[i][3];
  }
  __syncthreads();

  const int KT = K >> 5;
  for (int kt = 0; kt < KT; ++kt) {
    const int cur = kt & 1, nxt = cur ^ 1;
    const bool more = (kt + 1 < KT);
    if (more) {  // issue next tile's global loads before compute
      const int k0 = (kt + 1) << 5;
      ast.fetch(A, m0, K, k0, tid);
#pragma unroll
      for (int i = 0; i < 4; ++i) {
        int flat = i * 1024 + tid * 4;
        wr[i] = *(const v4f*)(W + (size_t)(k0 + (flat >> 7)) * N + n0 + (flat & 127));
      }
    }
    // ---- WMMA on current buffer (loads in flight above)
    v16bf af[2];
#pragma unroll
    for (int ms = 0; ms < 2; ++ms) {
      int row = wm * 32 + ms * 16 + nlo;
      af[ms] = cat16(*(const v8bf*)&sA[cur][row * 32 + kb],
                     *(const v8bf*)&sA[cur][row * 32 + kb + 16]);
    }
#pragma unroll
    for (int ns = 0; ns < 4; ++ns) {
      int col = wn * 64 + ns * 16 + nlo;
      v16bf bfr = cat16(*(const v8bf*)&sBT[cur][col * 32 + kb],
                        *(const v8bf*)&sBT[cur][col * 32 + kb + 16]);
#pragma unroll
      for (int ms = 0; ms < 2; ++ms)
        acc[ms][ns] = WMMA_BF16(af[ms], bfr, acc[ms][ns]);
    }
    if (more) {  // now drain the fetched regs into the other buffer
      ast.store(sA[nxt], tid);
#pragma unroll
      for (int i = 0; i < 4; ++i) {
        int flat = i * 1024 + tid * 4;
        int nn = flat & 127, kk = flat >> 7;
        sBT[nxt][(nn + 0) * 32 + kk] = (__bf16)wr[i][0];
        sBT[nxt][(nn + 1) * 32 + kk] = (__bf16)wr[i][1];
        sBT[nxt][(nn + 2) * 32 + kk] = (__bf16)wr[i][2];
        sBT[nxt][(nn + 3) * 32 + kk] = (__bf16)wr[i][3];
      }
    }
    __syncthreads();
  }

  // ---- epilogue: C layout (lane -> col, vgpr e -> row), add bias[col]
#pragma unroll
  for (int ms = 0; ms < 2; ++ms) {
#pragma unroll
    for (int ns = 0; ns < 4; ++ns) {
      int col = n0 + wn * 64 + ns * 16 + nlo;
      float bv = bias[col];
      int mbase = m0 + wm * 32 + ms * 16 + ((lane & 16) ? 8 : 0);
#pragma unroll
      for (int e = 0; e < 8; ++e)
        C[(size_t)(mbase + e) * N + col] = cvt_out<TC>(acc[ms][ns][e] + bv);
    }
  }
}

// ---------------------------------------------------------------------------
// Flash attention on bf16 qkv. Grid (T/64, B*H), 128 threads = 4 waves.
// Wave w owns query rows q0+16w..+15.  Q (64x64) and each K tile (32x64) are
// staged by the Tensor Data Mover (wave 0 issues, waits TENSORcnt, barrier
// publishes); V^T is transpose-scattered by all threads meanwhile.
// Row sums of P come from a ones-WMMA (result pre-broadcast across lanes),
// so only the row-max needs shuffle butterflies.
// ---------------------------------------------------------------------------
#define NH   12
#define TSEQ 2048
#define CDIM 768
#define QLD  2304

__global__ __launch_bounds__(128) void attn_kernel(const __bf16* __restrict__ qkv,
                                                   __bf16* __restrict__ y) {
  __shared__ __align__(128) __bf16 sQ[64 * 64];      // [qrow][d]
  __shared__ __align__(128) __bf16 sK[32 * 64];      // [key][d]
  __shared__ __align__(128) __bf16 sVT[64 * 32];     // [d][key]
  __shared__ __align__(128) __bf16 sP[4][16 * 32];   // per-wave P

  const int tid  = threadIdx.x;
  const int lane = tid & 31;
  const int wid  = tid >> 5;
  const int b    = blockIdx.y / NH;
  const int h    = blockIdx.y % NH;
  const int q0   = blockIdx.x * 64;
  const size_t base = (size_t)b * TSEQ * QLD;
  const int qoff = h * 64, koff = CDIM + h * 64, voff = 2 * CDIM + h * 64;
  const int kb  = (lane >> 4) * 8;
  const int nlo = lane & 15;
  const int mhi = (lane & 16) ? 8 : 0;

  // ---- stage Q block via TDM (64x64 bf16 tile, row stride QLD)
  if (wid == 0) {
    tdm_load_2d_bf16((unsigned int)(uintptr_t)sQ,
                     qkv + base + (size_t)q0 * QLD + qoff,
                     64, 64, QLD, 64, 64);
    __builtin_amdgcn_s_wait_tensorcnt(0);
  }
  __syncthreads();

  v16bf qf[2];
#pragma unroll
  for (int c = 0; c < 2; ++c) {
    int row = wid * 16 + nlo;
    qf[c] = cat16(*(const v8bf*)&sQ[row * 64 + c * 32 + kb],
                  *(const v8bf*)&sQ[row * 64 + c * 32 + kb + 16]);
  }

  v16bf onesf;                       // all-ones B fragment for row sums
#pragma unroll
  for (int i = 0; i < 16; ++i) onesf[i] = (__bf16)1.0f;
  const v8f zc = {};

  float mi[8], li[8];
#pragma unroll
  for (int e = 0; e < 8; ++e) { mi[e] = -3.0e30f; li[e] = 0.0f; }
  v8f o[4] = {};

  const int jend   = q0 + 64;
  const int rowmin = q0 + wid * 16;       // first row this wave owns
  const int rowlim = rowmin + 15;         // last row this wave owns

  for (int j0 = 0; j0 < jend; j0 += 32) {
    __syncthreads();  // everyone done reading previous K/V tile
    // ---- TDM copy of K tile overlaps manual V^T transpose-scatter
    if (wid == 0)
      tdm_load_2d_bf16((unsigned int)(uintptr_t)sK,
                       qkv + base + (size_t)j0 * QLD + koff,
                       64, 32, QLD, 64, 32);
#pragma unroll
    for (int i = 0; i < 2; ++i) {
      int flat = i * 1024 + tid * 8;
      int key = flat >> 6, d = flat & 63;
      v8bf vv = *(const v8bf*)(qkv + base + (size_t)(j0 + key) * QLD + voff + d);
#pragma unroll
      for (int j = 0; j < 8; ++j) sVT[(d + j) * 32 + key] = vv[j];
    }
    if (wid == 0) __builtin_amdgcn_s_wait_tensorcnt(0);
    __syncthreads();

    if (j0 <= rowlim) {  // tile not fully masked for this wave
      // ---- S = Q K^T : two 16x16 tiles over hd=64 (2 K-chunks)
      v8f s0 = {}, s1 = {};
#pragma unroll
      for (int c = 0; c < 2; ++c) {
        v16bf b0 = cat16(*(const v8bf*)&sK[nlo * 64 + c * 32 + kb],
                         *(const v8bf*)&sK[nlo * 64 + c * 32 + kb + 16]);
        s0 = WMMA_BF16(qf[c], b0, s0);
        v16bf b1 = cat16(*(const v8bf*)&sK[(16 + nlo) * 64 + c * 32 + kb],
                         *(const v8bf*)&sK[(16 + nlo) * 64 + c * 32 + kb + 16]);
        s1 = WMMA_BF16(qf[c], b1, s1);
      }

      // ---- online softmax (rows spread across 16-lane halves)
      const float scale = 0.125f;  // 1/sqrt(64)
      const bool mask_tile = (j0 + 31 > rowmin);  // touches the diagonal?
      float alpha[8];
#pragma unroll
      for (int e = 0; e < 8; ++e) {
        float v0 = s0[e] * scale;
        float v1 = s1[e] * scale;
        if (mask_tile) {  // only diagonal tiles pay the cndmask cost
          int qr = rowmin + mhi + e;
          if (j0 + nlo      > qr) v0 = -3.0e30f;
          if (j0 + 16 + nlo > qr) v1 = -3.0e30f;
        }
        float t = fmaxf(v0, v1);
#pragma unroll
        for (int off = 1; off < 16; off <<= 1) t = fmaxf(t, __shfl_xor(t, off, 16));
        float mn = fmaxf(mi[e], t);
        alpha[e] = __expf(mi[e] - mn);
        mi[e] = mn;
        float p0 = __expf(v0 - mn);
        float p1 = __expf(v1 - mn);
        int m = mhi + e;
        sP[wid][m * 32 + nlo]      = (__bf16)p0;  // C-layout -> A-layout via LDS
        sP[wid][m * 32 + 16 + nlo] = (__bf16)p1;
      }
#pragma unroll
      for (int nd = 0; nd < 4; ++nd)
#pragma unroll
        for (int e = 0; e < 8; ++e) o[nd][e] *= alpha[e];

      // same-wave DS is in-order; make the RAW explicit before the gather
      asm volatile("s_wait_dscnt 0x0" ::: "memory");

      // ---- P fragment; row sums via ones-WMMA (broadcast across lanes)
      v16bf pf = cat16(*(const v8bf*)&sP[wid][nlo * 32 + kb],
                       *(const v8bf*)&sP[wid][nlo * 32 + kb + 16]);
      v8f sl = WMMA_BF16(pf, onesf, zc);  // D(m,n) = rowsum(m) for every n
#pragma unroll
      for (int e = 0; e < 8; ++e) li[e] = li[e] * alpha[e] + sl[e];

      // ---- O += P V  (V^T supplies 4 B n-subtiles)
#pragma unroll
      for (int nd = 0; nd < 4; ++nd) {
        v16bf vf = cat16(*(const v8bf*)&sVT[(nd * 16 + nlo) * 32 + kb],
                         *(const v8bf*)&sVT[(nd * 16 + nlo) * 32 + kb + 16]);
        o[nd] = WMMA_BF16(pf, vf, o[nd]);
      }
    }
  }

  // ---- normalize, store O (16x64) as bf16 into yatt[B,T,C] head slice
  float inv[8];
#pragma unroll
  for (int e = 0; e < 8; ++e) inv[e] = 1.0f / li[e];
#pragma unroll
  for (int nd = 0; nd < 4; ++nd) {
    int col = h * 64 + nd * 16 + nlo;
#pragma unroll
    for (int e = 0; e < 8; ++e) {
      int qr = rowmin + mhi + e;
      y[((size_t)b * TSEQ + qr) * CDIM + col] = (__bf16)(o[nd][e] * inv[e]);
    }
  }
}

// ---------------------------------------------------------------------------
// Launch. Inputs (fp32): x, w_attn, b_attn, w_proj, b_proj (+2 scalars).
// ws layout: qkv bf16 [8192 x 2304] then yatt bf16 [8192 x 768] (~50 MB).
// ---------------------------------------------------------------------------
extern "C" void kernel_launch(void* const* d_in, const int* in_sizes, int n_in,
                              void* d_out, int out_size, void* d_ws, size_t ws_size,
                              hipStream_t stream) {
  (void)in_sizes; (void)n_in; (void)out_size; (void)ws_size;
  const float* x      = (const float*)d_in[0];
  const float* w_attn = (const float*)d_in[1];
  const float* b_attn = (const float*)d_in[2];
  const float* w_proj = (const float*)d_in[3];
  const float* b_proj = (const float*)d_in[4];
  float* y = (float*)d_out;

  const int BT = 4 * TSEQ;  // 8192 rows
  __bf16* qkv  = (__bf16*)d_ws;
  __bf16* yatt = (__bf16*)((char*)d_ws + (size_t)BT * QLD * sizeof(__bf16));

  // 1) qkv(bf16) = x @ w_attn + b_attn   (M=8192, N=2304, K=768)
  dim3 g1(QLD / 128, BT / 128);
  gemm_bias_kernel<float, __bf16><<<g1, 256, 0, stream>>>(x, w_attn, b_attn, qkv, BT, QLD, CDIM);

  // 2) flash attention -> yatt(bf16)
  dim3 g2(TSEQ / 64, 4 * NH);
  attn_kernel<<<g2, 128, 0, stream>>>(qkv, yatt);

  // 3) y(fp32) = yatt @ w_proj + b_proj  (M=8192, N=768, K=768)
  dim3 g3(CDIM / 128, BT / 128);
  gemm_bias_kernel<__bf16, float><<<g3, 256, 0, stream>>>(yatt, w_proj, b_proj, y, BT, CDIM, CDIM);
}
